// DepthAwareGATv2Backbone_15522011808342
// MI455X (gfx1250) — matrix operand order repaired and checked by reading, not compiled, not run on hardware
//
#include <hip/hip_runtime.h>
#include <math.h>

typedef __attribute__((ext_vector_type(16))) _Float16 v16h;
typedef __attribute__((ext_vector_type(8)))  float    v8f;

#define LRELU 0.2f
#define LNEPS 1e-5f
#define WSTR 132   // padded LDS row stride (halves): staggers banks, keeps 32b pair alignment

// ---------------------------------------------------------------- fill
__global__ void fillf(float* __restrict__ p, float v, int n) {
  int i = blockIdx.x * blockDim.x + threadIdx.x;
  if (i < n) p[i] = v;
}

// ---------------------------------------------------------------- GEMM
// Y[nrows x 128] = X[nrows x K](f32, row stride ldx) * W^T  (W: [128 x K] f32) + bias
// 256 threads = 8 waves; block tile 128 rows x 128 cols; wave tile 16 x 128.
// W chunk (128x128) staged in LDS as f16; A fragments converted f32->f16 in regs.
__global__ void __launch_bounds__(256)
gemm_xwt(const float* __restrict__ X, int ldx,
         const float* __restrict__ W, int K,
         const float* __restrict__ bias,
         float* __restrict__ Y, int ldy, int nrows)
{
  __shared__ _Float16 wl[128 * WSTR];
  const int tid  = threadIdx.x;
  const int wave = tid >> 5;
  const int lane = tid & 31;
  const int r  = lane & 15;
  const int hh = lane >> 4;
  const int rowBase = blockIdx.x * 128 + wave * 16;
  const int arow = rowBase + r;

  v8f acc[8] = {};

  for (int kc = 0; kc < K; kc += 128) {
    // cooperative stage of W[:, kc:kc+128] -> LDS (f16), float4 loads
    for (int i = tid; i < 128 * 32; i += 256) {
      const int d = i >> 5;
      const int k = (i & 31) << 2;
      const float4 wv = *reinterpret_cast<const float4*>(W + (size_t)d * K + kc + k);
      _Float16* dp = wl + d * WSTR + k;
      dp[0] = (_Float16)wv.x; dp[1] = (_Float16)wv.y;
      dp[2] = (_Float16)wv.z; dp[3] = (_Float16)wv.w;
    }
    __syncthreads();

    for (int k0 = 0; k0 < 128; k0 += 32) {
      // A fragment 16x32 (contiguous k, matches B): lanes 0-15: K=0..15;
      // lanes 16-31: K=16..31
      v16h a;
      if (arow < nrows) {
        const float* xp = X + (size_t)arow * ldx + kc + k0;
        #pragma unroll
        for (int v = 0; v < 4; ++v) {
          const int ka = hh * 16 + 2 * v;
          a[2*v]     = (_Float16)xp[ka];
          a[2*v+1]   = (_Float16)xp[ka + 1];
          const int kb = hh * 16 + 8 + 2 * v;
          a[8+2*v]   = (_Float16)xp[kb];
          a[8+2*v+1] = (_Float16)xp[kb + 1];
        }
      } else {
        #pragma unroll
        for (int v = 0; v < 16; ++v) a[v] = (_Float16)0.f;
      }
      // 8 col tiles: B fragment 32x16: lane = col, lanes 0-15 K=0..15, 16-31 K=16..31
      #pragma unroll
      for (int j = 0; j < 8; ++j) {
        const _Float16* wp = wl + (j * 16 + r) * WSTR + k0 + hh * 16;
        v16h b;
        #pragma unroll
        for (int v = 0; v < 16; ++v) b[v] = wp[v];
        acc[j] = __builtin_amdgcn_wmma_f32_16x16x32_f16(
            false, a, false, b, (short)0, acc[j], false, false);
      }
    }
    __syncthreads();
  }

  // D layout: lane n = lane&15, VGPR v -> row hh*8+v
  #pragma unroll
  for (int j = 0; j < 8; ++j) {
    const int col = j * 16 + r;
    const float bv = bias ? bias[col] : 0.f;
    #pragma unroll
    for (int v = 0; v < 8; ++v) {
      const int row = rowBase + hh * 8 + v;
      if (row < nrows) Y[(size_t)row * ldy + col] = acc[j][v] + bv;
    }
  }
}

// ------------------------------------------------- edge features + encoder
// wave per edge: float4 gather of x[src],x[dst], wave-reduce 4 polynomial feats,
// MLP 4->16(relu)->8 via lane shuffles; block-reduced running sum for mean fill.
__global__ void __launch_bounds__(256)
edge_feat_enc(const float* __restrict__ x, const int* __restrict__ srcv,
              const int* __restrict__ dstv,
              const float* __restrict__ w1, const float* __restrict__ b1,
              const float* __restrict__ w2, const float* __restrict__ b2,
              float* __restrict__ ea, float* __restrict__ easum, int E)
{
  __shared__ float sea[8][8];
  const int lane = threadIdx.x & 31;
  const int wave = threadIdx.x >> 5;
  const int e = blockIdx.x * 8 + wave;
  const bool active = (e < E);

  float dotv = 0.f, sqd = 0.f, ni = 0.f, nj = 0.f;
  if (active) {
    const int s = srcv[e], d = dstv[e];
    const float4 xi = *reinterpret_cast<const float4*>(x + (size_t)s * 128 + lane * 4);
    const float4 xj = *reinterpret_cast<const float4*>(x + (size_t)d * 128 + lane * 4);
    dotv = xi.x*xj.x + xi.y*xj.y + xi.z*xj.z + xi.w*xj.w;
    const float dx = xi.x-xj.x, dy = xi.y-xj.y, dz = xi.z-xj.z, dw = xi.w-xj.w;
    sqd = dx*dx + dy*dy + dz*dz + dw*dw;
    ni  = xi.x*xi.x + xi.y*xi.y + xi.z*xi.z + xi.w*xi.w;
    nj  = xj.x*xj.x + xj.y*xj.y + xj.z*xj.z + xj.w*xj.w;
  }
  #pragma unroll
  for (int m = 16; m >= 1; m >>= 1) {
    dotv += __shfl_xor(dotv, m, 32);
    sqd  += __shfl_xor(sqd,  m, 32);
    ni   += __shfl_xor(ni,   m, 32);
    nj   += __shfl_xor(nj,   m, 32);
  }
  float hid = 0.f;
  if (lane < 16)
    hid = fmaxf(b1[lane] + w1[lane*4]*dotv + w1[lane*4+1]*sqd
                + w1[lane*4+2]*ni + w1[lane*4+3]*nj, 0.f);
  float o = (lane < 8) ? b2[lane] : 0.f;
  #pragma unroll
  for (int l = 0; l < 16; ++l) {
    const float hv = __shfl(hid, l, 32);
    if (lane < 8) o += w2[lane*16 + l] * hv;
  }
  if (lane < 8) {
    if (active) ea[(size_t)e*8 + lane] = o;
    sea[wave][lane] = active ? o : 0.f;
  }
  __syncthreads();
  if (threadIdx.x < 8) {
    float s = 0.f;
    #pragma unroll
    for (int wv = 0; wv < 8; ++wv) s += sea[wv][threadIdx.x];
    atomicAdd(easum + threadIdx.x, s);
  }
}

// ------------------------------------------------- tiny prep: ea mean + M = att_edge-folded lin_edge_w
// a_edge[e][h] = sum_c (ea @ lin_edge_w^T)[h*16+c] * att_edge[h][c] = sum_h' ea[h'] * M[h][h']
__global__ void prep_small(const float* __restrict__ easum, float* __restrict__ eamean, int E,
                           const float* __restrict__ lew1, const float* __restrict__ ae1,
                           float* __restrict__ M1,
                           const float* __restrict__ lew2, const float* __restrict__ ae2,
                           float* __restrict__ M2)
{
  const int t = threadIdx.x;
  if (t < 8) eamean[t] = easum[t] / (float)E;
  if (t < 64) {
    const int h = t >> 3, hp = t & 7;
    float s1 = 0.f, s2 = 0.f;
    #pragma unroll
    for (int c = 0; c < 16; ++c) {
      s1 += lew1[(h*16 + c)*8 + hp] * ae1[h*16 + c];
      s2 += lew2[(h*16 + c)*8 + hp] * ae2[h*16 + c];
    }
    M1[t] = s1; M2[t] = s2;
  }
}

// ------------------------------------------------- per-(node,head) attention scores
__global__ void node_scores(const float* __restrict__ xs,
                            const float* __restrict__ att_src,
                            const float* __restrict__ att_dst,
                            float* __restrict__ asrc, float* __restrict__ adst, int N)
{
  const int idx = blockIdx.x * blockDim.x + threadIdx.x;
  if (idx >= N * 8) return;
  const int n = idx >> 3, h = idx & 7;
  const float* xp = xs + (size_t)n * 128 + h * 16;
  float s1 = 0.f, s2 = 0.f;
  #pragma unroll
  for (int c = 0; c < 16; ++c) {
    const float v = xp[c];
    s1 += v * att_src[h*16 + c];
    s2 += v * att_dst[h*16 + c];
  }
  asrc[idx] = s1; adst[idx] = s2;
}

// ------------------------------------------------- edge logits + segment max
__global__ void edge_logits(const float* __restrict__ ea, const float* __restrict__ eamean,
                            const float* __restrict__ Mmat,
                            const int* __restrict__ srcv, const int* __restrict__ dstv,
                            const float* __restrict__ asrc, const float* __restrict__ adst,
                            float* __restrict__ wbuf, float* __restrict__ mx, int E, int N)
{
  const int e = blockIdx.x * blockDim.x + threadIdx.x;
  if (e >= E + N) return;
  int s, d;
  float ef[8];
  if (e < E) {
    s = srcv[e]; d = dstv[e];
    #pragma unroll
    for (int h = 0; h < 8; ++h) ef[h] = ea[(size_t)e*8 + h];
  } else {
    s = d = e - E;
    #pragma unroll
    for (int h = 0; h < 8; ++h) ef[h] = eamean[h];
  }
  #pragma unroll
  for (int h = 0; h < 8; ++h) {
    float aev = 0.f;
    #pragma unroll
    for (int hp = 0; hp < 8; ++hp) aev += ef[hp] * Mmat[h*8 + hp];
    float al = asrc[s*8 + h] + adst[d*8 + h] + aev;
    al = (al > 0.f) ? al : LRELU * al;
    wbuf[(size_t)e*8 + h] = al;
    __hip_atomic_fetch_max(mx + d*8 + h, al, __ATOMIC_RELAXED, __HIP_MEMORY_SCOPE_AGENT);
  }
}

// ------------------------------------------------- exp + segment denom
__global__ void edge_softmax_num(const int* __restrict__ dstv,
                                 float* __restrict__ wbuf, const float* __restrict__ mx,
                                 float* __restrict__ denom, int E, int N)
{
  const int e = blockIdx.x * blockDim.x + threadIdx.x;
  if (e >= E + N) return;
  const int d = (e < E) ? dstv[e] : (e - E);
  #pragma unroll
  for (int h = 0; h < 8; ++h) {
    const float wv = __expf(wbuf[(size_t)e*8 + h] - mx[d*8 + h]);
    wbuf[(size_t)e*8 + h] = wv;
    atomicAdd(denom + d*8 + h, wv);
  }
}

// ------------------------------------------------- scatter-aggregate (wave per edge)
__global__ void __launch_bounds__(256)
edge_aggregate(const int* __restrict__ srcv, const int* __restrict__ dstv,
               const float* __restrict__ xs, const float* __restrict__ wbuf,
               const float* __restrict__ denom, float* __restrict__ agg, int E, int N)
{
  const int lane = threadIdx.x & 31;
  const int wave = threadIdx.x >> 5;
  const int e = blockIdx.x * 8 + wave;
  if (e >= E + N) return;
  int s, d;
  if (e < E) { s = srcv[e]; d = dstv[e]; } else { s = d = e - E; }
  const int h = lane >> 2;                      // 4 channels per lane, all same head
  const float coef = wbuf[(size_t)e*8 + h] / (denom[d*8 + h] + 1e-16f);
  const float4 xv = *reinterpret_cast<const float4*>(xs + (size_t)s*128 + lane*4);
  float* ap = agg + (size_t)d*128 + lane*4;
  atomicAdd(ap + 0, xv.x * coef);
  atomicAdd(ap + 1, xv.y * coef);
  atomicAdd(ap + 2, xv.z * coef);
  atomicAdd(ap + 3, xv.w * coef);
}

// ------------------------------------------------- fused bias + ELU + residual + LayerNorm
__global__ void __launch_bounds__(256)
finalize_ln(const float* __restrict__ agg, const float* __restrict__ bias,
            const float* __restrict__ res, int res_ld,
            const float* __restrict__ g, const float* __restrict__ bv,
            float* __restrict__ outp, int out_ld, int N)
{
  const int lane = threadIdx.x & 31;
  const int wave = threadIdx.x >> 5;
  const int n = blockIdx.x * 8 + wave;
  if (n >= N) return;
  const int i0 = lane * 4;
  float t[4];
  float ssum = 0.f, ssq = 0.f;
  #pragma unroll
  for (int j = 0; j < 4; ++j) {
    float v = agg[(size_t)n*128 + i0 + j] + bias[i0 + j];
    v = (v > 0.f) ? v : (__expf(v) - 1.f);                // ELU
    t[j] = v + res[(size_t)n*res_ld + i0 + j];            // residual
    ssum += t[j];
    ssq  += t[j] * t[j];
  }
  #pragma unroll
  for (int m = 16; m >= 1; m >>= 1) {
    ssum += __shfl_xor(ssum, m, 32);
    ssq  += __shfl_xor(ssq,  m, 32);
  }
  const float mu  = ssum * (1.f / 128.f);
  const float var = ssq  * (1.f / 128.f) - mu * mu;
  const float rs  = rsqrtf(var + LNEPS);
  #pragma unroll
  for (int j = 0; j < 4; ++j)
    outp[(size_t)n*out_ld + i0 + j] = (t[j] - mu) * rs * g[i0 + j] + bv[i0 + j];
}

// ================================================================ host side
static void gat_layer(const float* xs, const float* att_src, const float* att_dst,
                      const float* Mmat, const float* ea, const float* eamean,
                      const int* srcv, const int* dstv,
                      float* asrc, float* adst, float* mx, float* denom,
                      float* wbuf, float* agg,
                      const float* bias, const float* res, const float* g, const float* bvec,
                      float* outp, int N, int E, hipStream_t stream)
{
  const int EN = E + N;
  node_scores<<<(N*8 + 255)/256, 256, 0, stream>>>(xs, att_src, att_dst, asrc, adst, N);
  fillf<<<(N*8 + 255)/256, 256, 0, stream>>>(mx, -3.0e38f, N*8);
  fillf<<<(N*8 + 255)/256, 256, 0, stream>>>(denom, 0.f, N*8);
  fillf<<<(N*128 + 255)/256, 256, 0, stream>>>(agg, 0.f, N*128);
  edge_logits<<<(EN + 255)/256, 256, 0, stream>>>(ea, eamean, Mmat, srcv, dstv,
                                                  asrc, adst, wbuf, mx, E, N);
  edge_softmax_num<<<(EN + 255)/256, 256, 0, stream>>>(dstv, wbuf, mx, denom, E, N);
  edge_aggregate<<<(EN + 7)/8, 256, 0, stream>>>(srcv, dstv, xs, wbuf, denom, agg, E, N);
  finalize_ln<<<(N + 7)/8, 256, 0, stream>>>(agg, bias, res, 384, g, bvec, outp, 384, N);
}

extern "C" void kernel_launch(void* const* d_in, const int* in_sizes, int n_in,
                              void* d_out, int out_size, void* d_ws, size_t ws_size,
                              hipStream_t stream)
{
  const float* x           = (const float*)d_in[0];
  const int*   ei          = (const int*)  d_in[1];
  const float* enc_w1      = (const float*)d_in[2];
  const float* enc_b1      = (const float*)d_in[3];
  const float* enc_w2      = (const float*)d_in[4];
  const float* enc_b2      = (const float*)d_in[5];
  const float* in_w        = (const float*)d_in[6];
  const float* in_b        = (const float*)d_in[7];
  const float* lin_w1      = (const float*)d_in[8];
  const float* att_src1    = (const float*)d_in[9];
  const float* att_dst1    = (const float*)d_in[10];
  const float* att_edge1   = (const float*)d_in[11];
  const float* lin_edge_w1 = (const float*)d_in[12];
  const float* bias1       = (const float*)d_in[13];
  const float* lin_w2      = (const float*)d_in[14];
  const float* att_src2    = (const float*)d_in[15];
  const float* att_dst2    = (const float*)d_in[16];
  const float* att_edge2   = (const float*)d_in[17];
  const float* lin_edge_w2 = (const float*)d_in[18];
  const float* bias2       = (const float*)d_in[19];
  const float* g1          = (const float*)d_in[20];
  const float* bn1         = (const float*)d_in[21];
  const float* g2          = (const float*)d_in[22];
  const float* bn2         = (const float*)d_in[23];
  const float* jk_w        = (const float*)d_in[24];
  const float* jk_b        = (const float*)d_in[25];

  const int N  = in_sizes[0] / 128;
  const int E  = in_sizes[1] / 2;
  const int* srcv = ei;
  const int* dstv = ei + E;

  // workspace layout (all float, 16B-aligned segments)
  float* w = (float*)d_ws;
  float* ea     = w; w += (size_t)E * 8;
  float* easum  = w; w += 8;
  float* eamean = w; w += 8;
  float* M1     = w; w += 64;
  float* M2     = w; w += 64;
  float* asrc   = w; w += (size_t)N * 8;
  float* adst   = w; w += (size_t)N * 8;
  float* mx     = w; w += (size_t)N * 8;
  float* denom  = w; w += (size_t)N * 8;
  float* wbuf   = w; w += (size_t)(E + N) * 8;
  float* xs     = w; w += (size_t)N * 128;
  float* agg    = w; w += (size_t)N * 128;
  float* hcat   = w; w += (size_t)N * 384;   // [h0 | h1 | h2]
  (void)ws_size; (void)n_in; (void)out_size;

  // edge encoder + mean + folded edge-attention matrices
  fillf<<<1, 32, 0, stream>>>(easum, 0.f, 8);
  edge_feat_enc<<<(E + 7)/8, 256, 0, stream>>>(x, srcv, dstv, enc_w1, enc_b1,
                                               enc_w2, enc_b2, ea, easum, E);
  prep_small<<<1, 64, 0, stream>>>(easum, eamean, E,
                                   lin_edge_w1, att_edge1, M1,
                                   lin_edge_w2, att_edge2, M2);

  const int gblocks = (N + 127) / 128;
  // h0 = x @ in_w^T + in_b  -> hcat[:, 0:128]
  gemm_xwt<<<gblocks, 256, 0, stream>>>(x, 128, in_w, 128, in_b, hcat, 384, N);
  // xs1 = x @ lin_w1^T
  gemm_xwt<<<gblocks, 256, 0, stream>>>(x, 128, lin_w1, 128, nullptr, xs, 128, N);

  // conv1 -> h1 = LN(elu(gat)+h0) -> hcat[:, 128:256]
  gat_layer(xs, att_src1, att_dst1, M1, ea, eamean, srcv, dstv,
            asrc, adst, mx, denom, wbuf, agg,
            bias1, hcat, g1, bn1, hcat + 128, N, E, stream);

  // xs2 = h1 @ lin_w2^T   (h1 lives at hcat+128 with row stride 384)
  gemm_xwt<<<gblocks, 256, 0, stream>>>(hcat + 128, 384, lin_w2, 128, nullptr, xs, 128, N);

  // conv2 -> h2 = LN(elu(gat)+h1) -> hcat[:, 256:384]
  gat_layer(xs, att_src2, att_dst2, M2, ea, eamean, srcv, dstv,
            asrc, adst, mx, denom, wbuf, agg,
            bias2, hcat + 128, g2, bn2, hcat + 256, N, E, stream);

  // out = [h0|h1|h2] @ jk_w^T + jk_b
  gemm_xwt<<<gblocks, 256, 0, stream>>>(hcat, 384, jk_w, 384, jk_b,
                                        (float*)d_out, 128, N);
}
